// MultiViewTGCN_85959475462400
// MI455X (gfx1250) — matrix operand order repaired and checked by reading, not compiled, not run on hardware
//
#include <hip/hip_runtime.h>

typedef __attribute__((ext_vector_type(16))) __bf16 bf16x16;
typedef __attribute__((ext_vector_type(8)))  float  f32x8;
typedef int i32x4v __attribute__((vector_size(16)));

union FragU { uint4 q[2]; bf16x16 v; unsigned short u[16]; unsigned p[8]; };

#if __has_builtin(__builtin_amdgcn_global_load_async_to_lds_b128)
#define HAVE_ASYNC_LDS 1
#else
#define HAVE_ASYNC_LDS 0
#endif

__device__ __forceinline__ unsigned short f2bf(float f) {
  unsigned u = __float_as_uint(f);
  u += 0x7FFFu + ((u >> 16) & 1u);          // round-to-nearest-even
  return (unsigned short)(u >> 16);
}

// Packed RNE pair conversion: lo -> [15:0], hi -> [31:16].
// 2x (v_bfe+v_add3) + one v_perm_b32 byte-select = 5 VALU per pair.
__device__ __forceinline__ unsigned f2bf2(float lo, float hi) {
#if __has_builtin(__builtin_amdgcn_cvt_pk_bf16_f32)
  auto r = __builtin_amdgcn_cvt_pk_bf16_f32(lo, hi);   // v_cvt_pk_bf16_f32
  return __builtin_bit_cast(unsigned, r);
#elif __has_builtin(__builtin_amdgcn_perm)
  unsigned u0 = __float_as_uint(lo);
  unsigned u1 = __float_as_uint(hi);
  u0 += 0x7FFFu + ((u0 >> 16) & 1u);
  u1 += 0x7FFFu + ((u1 >> 16) & 1u);
  // bytes {u1[3],u1[2],u0[3],u0[2]} -> {hi_bf16, lo_bf16}
  return __builtin_amdgcn_perm(u1, u0, 0x07060302u);
#else
  return (unsigned)f2bf(lo) | ((unsigned)f2bf(hi) << 16);
#endif
}

__device__ __forceinline__ void wait_async0() {
#if __has_builtin(__builtin_amdgcn_s_wait_asynccnt)
  __builtin_amdgcn_s_wait_asynccnt(0);
#else
  asm volatile("s_wait_asynccnt 0x0" ::: "memory");
#endif
}

__device__ __forceinline__ bf16x16 frag_zero() {
  FragU f; f.q[0] = make_uint4(0,0,0,0); f.q[1] = make_uint4(0,0,0,0); return f.v;
}

// A fragment: 16x32 bf16 tile, src row-major [ld], M across lanes (mod 16).
// lane<16 holds K elems 0..15; lane>=16 holds K 16..31.
__device__ __forceinline__ bf16x16 frag_a(const unsigned short* src, int ld,
                                          int rbase, int kbase, int lane) {
  const unsigned short* p = src + (rbase + (lane & 15)) * ld + kbase;
  const int c = (lane >> 4) * 16;
  FragU f;
  f.q[0] = *(const uint4*)(p + c);
  f.q[1] = *(const uint4*)(p + c + 8);
  return f.v;
}

// B fragment: 32x16 bf16 tile from row-major weights [g][ld] (g = output col).
// lane -> col N = lane&15; lane<16 holds K 0..15, lane>=16 holds K 16..31.
__device__ __forceinline__ bf16x16 frag_b(const unsigned short* src, int ld,
                                          int gbase, int kbase, int lane) {
  const unsigned short* p = src + (gbase + (lane & 15)) * ld + kbase + ((lane >> 4) * 16);
  FragU f;
  f.q[0] = *(const uint4*)(p);
  f.q[1] = *(const uint4*)(p + 8);
  return f.v;
}

// Same as frag_b but converts from an f32 row-major weight matrix on the fly.
__device__ __forceinline__ bf16x16 frag_b_f32(const float* src, int ld,
                                              int gbase, int kbase, int lane) {
  const float* p = src + (gbase + (lane & 15)) * ld + kbase + ((lane >> 4) * 16);
  FragU f;
  #pragma unroll
  for (int i = 0; i < 8; ++i) f.p[i] = f2bf2(p[2 * i], p[2 * i + 1]);
  return f.v;
}

__device__ __forceinline__ f32x8 wmma_bf16(bf16x16 a, bf16x16 b, f32x8 c) {
  return __builtin_amdgcn_wmma_f32_16x16x32_bf16(false, a, false, b, (short)0, c,
                                                 false, false);
}

// ---------------------------------------------------------------------------
// Kernel 1: per-(view,timestep) GCN + ReLU, writes h_seq in bf16 GRU layout.
// h_seq[(l*8+b)*4096 + n*128 + v*32 + h]
// ---------------------------------------------------------------------------
__global__ void gcn_kernel(const float* __restrict__ x,
                           const int* __restrict__ eiA, const int* __restrict__ eiB,
                           const int* __restrict__ eiC, const int* __restrict__ eiD,
                           const float* __restrict__ Wg, const float* __restrict__ bg,
                           unsigned short* __restrict__ hseq) {
  constexpr int L = 24, Nn = 32, F = 16, HC = 32, E = 256;
  __shared__ float sW[F * HC];          // 2 KB
  __shared__ float sXW[256 * HC];       // 32 KB
  __shared__ float sAgg[Nn * HC];       // 4 KB
  __shared__ float sDeg[Nn];
  __shared__ float sDinv[Nn];

  const int tid = threadIdx.x;
  const int v = blockIdx.x / L;
  const int l = blockIdx.x % L;
  const int* ei = (v == 0) ? eiA : (v == 1) ? eiB : (v == 2) ? eiC : eiD;

  for (int i = tid; i < F * HC; i += 256) sW[i] = Wg[v * F * HC + i];
  for (int i = tid; i < Nn * HC; i += 256) sAgg[i] = 0.f;
  if (tid < Nn) sDeg[tid] = 1.f;
  __syncthreads();

  atomicAdd(&sDeg[ei[E + tid]], 1.f);   // in-degree of A+I (dst < N always)
  __syncthreads();

  if (tid < Nn) sDinv[tid] = rsqrtf(sDeg[tid]);

  // this thread owns node m = tid = b*32+n; load its feature row
  const int b = tid >> 5, n = tid & 31;
  float xr[F];
  const float* px = x + (((b * L) + l) * Nn + n) * F;
  #pragma unroll
  for (int f = 0; f < F; ++f) xr[f] = px[f];
  __syncthreads();

  #pragma unroll 4
  for (int h = 0; h < HC; ++h) {
    float s = 0.f;
    #pragma unroll
    for (int f = 0; f < F; ++f) s += xr[f] * sW[f * HC + h];
    sXW[tid * HC + h] = s;
  }
  __syncthreads();

  { // edge aggregation: thread owns channel h = tid&31, 32 edges
    const int h = tid & 31;
    const int eg = (tid >> 5) * 32;
    for (int k = 0; k < 32; ++k) {
      const int e = eg + k;
      const int s = ei[e], d = ei[E + e];
      const float norm = sDinv[s] * sDinv[d];
      atomicAdd(&sAgg[d * HC + h], sXW[s * HC + h] * norm);
    }
  }
  __syncthreads();

  const float invdeg = (tid < Nn) ? (1.f / sDeg[tid]) : 1.f;
  unsigned* po = (unsigned*)(hseq + (l * 8 + b) * 4096 + n * 128 + v * 32);
  #pragma unroll 4
  for (int h = 0; h < HC; h += 2) {
    float v0 = sXW[tid * HC + h] * invdeg + bg[v * HC + h];
    float v1 = sXW[tid * HC + h + 1] * invdeg + bg[v * HC + h + 1];
    if (tid < Nn) { v0 += sAgg[tid * HC + h]; v1 += sAgg[tid * HC + h + 1]; }
    po[h >> 1] = f2bf2(fmaxf(v0, 0.f), fmaxf(v1, 0.f));
  }
}

// ---------------------------------------------------------------------------
// Kernel 2: fp32 -> bf16 weight conversion (w_hh becomes L2-resident: 100 MB)
// ---------------------------------------------------------------------------
__global__ void cvt_bf16_kernel(const float* __restrict__ src,
                                unsigned short* __restrict__ dst, int n8) {
  const int i = blockIdx.x * blockDim.x + threadIdx.x;
  if (i >= n8) return;
  const float4* s = (const float4*)src + (size_t)i * 2;
  const float4 a = s[0], c = s[1];
  const uint4 o = make_uint4(f2bf2(a.x, a.y), f2bf2(a.z, a.w),
                             f2bf2(c.x, c.y), f2bf2(c.z, c.w));
  ((uint4*)dst)[i] = o;
}

// ---------------------------------------------------------------------------
// Kernel 3: xp = h_seq(192x4096,bf16) @ w_ih^T(4096x12288,f32->bf16) + b_ih
// 64x64 workgroup tile, LDS-staged, 8 waves x 2 WMMA tiles.
// A tile staged via async global->LDS copy (ASYNCcnt).
// ---------------------------------------------------------------------------
__global__ void xp_gemm_kernel(const unsigned short* __restrict__ hseq,
                               const float* __restrict__ w_ih,
                               const float* __restrict__ b_ih,
                               float* __restrict__ xp) {
  constexpr int K = 4096, GN = 12288;
  __shared__ __align__(16) unsigned short sA[64 * 32];
  __shared__ __align__(16) unsigned short sB[64 * 32];
  const int tid = threadIdx.x;
  const int lane = tid & 31, wave = tid >> 5;
  const int m_base = blockIdx.x * 64;
  const int n_base = blockIdx.y * 64;
  const int mi = wave & 3;
  const int ni = (wave >> 2) * 2;
  const int srow = tid >> 2;           // staging: 0..63
  const int scol = (tid & 3) * 8;      // staging: 0,8,16,24

  f32x8 acc0 = {}; f32x8 acc1 = {};
  for (int kb = 0; kb < K; kb += 32) {
    // stage A (pure bf16 copy): async DMA to LDS when the builtin exists
#if HAVE_ASYNC_LDS
    __builtin_amdgcn_global_load_async_to_lds_b128(
        (i32x4v*)(hseq + (m_base + srow) * K + kb + scol),
        (i32x4v*)(sA + srow * 32 + scol),
        0, 0);
#else
    *(uint4*)(sA + srow * 32 + scol) =
        *(const uint4*)(hseq + (m_base + srow) * K + kb + scol);
#endif
    { // stage B with fused f32 -> bf16 conversion (packed RNE + v_perm)
      const float* p = w_ih + (n_base + srow) * K + kb + scol;
      const float4 a = *(const float4*)p;
      const float4 c = *(const float4*)(p + 4);
      const uint4 o = make_uint4(f2bf2(a.x, a.y), f2bf2(a.z, a.w),
                                 f2bf2(c.x, c.y), f2bf2(c.z, c.w));
      *(uint4*)(sB + srow * 32 + scol) = o;
    }
#if HAVE_ASYNC_LDS
    wait_async0();
#endif
    __syncthreads();
    const bf16x16 af = frag_a(sA, 32, mi * 16, 0, lane);
    const bf16x16 b0 = frag_b(sB, 32, ni * 16, 0, lane);
    const bf16x16 b1 = frag_b(sB, 32, ni * 16 + 16, 0, lane);
    acc0 = wmma_bf16(af, b0, acc0);
    acc1 = wmma_bf16(af, b1, acc1);
    __syncthreads();
  }
  // D layout: col = lane&15, row = vgpr + 8*(lane>=16)
  const int mrow = m_base + mi * 16 + 8 * (lane >> 4);
  int j = n_base + ni * 16 + (lane & 15);
  float bias = b_ih[j];
  #pragma unroll
  for (int v2 = 0; v2 < 8; ++v2) xp[(mrow + v2) * GN + j] = acc0[v2] + bias;
  j += 16;
  bias = b_ih[j];
  #pragma unroll
  for (int v2 = 0; v2 < 8; ++v2) xp[(mrow + v2) * GN + j] = acc1[v2] + bias;
}

// ---------------------------------------------------------------------------
// Kernel 4: one GRU time step. 256 blocks x 3 waves (one per gate, same
// 16-column tile). Batch=8 -> A rows 8..15 zero-padded. Direct global frags
// (w_hh bf16 is L2-resident), cross-gate combine via LDS, fused gates.
// ---------------------------------------------------------------------------
__global__ void gru_step_kernel(const unsigned short* __restrict__ whh,
                                const float* __restrict__ xp,
                                const float* __restrict__ b_hh,
                                const float* __restrict__ hprev_f,
                                const unsigned short* __restrict__ hprev_b,
                                float* __restrict__ hnext_f,
                                unsigned short* __restrict__ hnext_b,
                                int t) {
  constexpr int H = 4096, GN = 12288;
  __shared__ float sHP[3][8][16];
  const int tid = threadIdx.x;
  const int lane = tid & 31;
  const int g = tid >> 5;                 // gate: 0=r 1=z 2=n
  const int j0 = blockIdx.x * 16;
  const int r = lane & 15;

  f32x8 acc = {};
  for (int kb = 0; kb < H; kb += 32) {
    const bf16x16 af = (r < 8) ? frag_a(hprev_b, H, 0, kb, lane) : frag_zero();
    const bf16x16 bf = frag_b(whh, H, g * H + j0, kb, lane);
    acc = wmma_bf16(af, bf, acc);
  }
  if (lane < 16) {                        // valid rows b=0..7 live in lanes 0..15
    #pragma unroll
    for (int v2 = 0; v2 < 8; ++v2) sHP[g][v2][lane] = acc[v2];
  }
  __syncthreads();

  for (int o = tid; o < 128; o += 96) {
    const int b = o >> 4, jj = o & 15, j = j0 + jj;
    const float* xrow = xp + (t * 8 + b) * GN;
    const float hr = sHP[0][b][jj] + b_hh[j];
    const float hz = sHP[1][b][jj] + b_hh[H + j];
    const float hn = sHP[2][b][jj] + b_hh[2 * H + j];
    const float rg = 1.f / (1.f + __expf(-(xrow[j] + hr)));
    const float zg = 1.f / (1.f + __expf(-(xrow[H + j] + hz)));
    const float ng = tanhf(xrow[2 * H + j] + rg * hn);
    const float hv = (1.f - zg) * ng + zg * hprev_f[b * H + j];
    hnext_f[b * H + j] = hv;
    hnext_b[b * H + j] = f2bf(hv);
  }
}

// ---------------------------------------------------------------------------
// Kernel 5: out = h_last(8x4096) @ fc_w^T(4096x128) + fc_b  (one workgroup)
// ---------------------------------------------------------------------------
__global__ void fc_kernel(const unsigned short* __restrict__ hb,
                          const float* __restrict__ fc_w,
                          const float* __restrict__ fc_b,
                          float* __restrict__ out) {
  constexpr int H = 4096;
  const int tid = threadIdx.x;
  const int lane = tid & 31, wave = tid >> 5;
  const int r = lane & 15;
  f32x8 acc = {};
  for (int kb = 0; kb < H; kb += 32) {
    const bf16x16 af = (r < 8) ? frag_a(hb, H, 0, kb, lane) : frag_zero();
    const bf16x16 bf = frag_b_f32(fc_w, H, wave * 16, kb, lane);
    acc = wmma_bf16(af, bf, acc);
  }
  if (lane < 16) {
    const int j = wave * 16 + lane;
    const float bias = fc_b[j];
    #pragma unroll
    for (int v2 = 0; v2 < 8; ++v2) out[v2 * 128 + j] = acc[v2] + bias;
  }
}

// ---------------------------------------------------------------------------
extern "C" void kernel_launch(void* const* d_in, const int* in_sizes, int n_in,
                              void* d_out, int out_size, void* d_ws, size_t ws_size,
                              hipStream_t stream) {
  const float* x    = (const float*)d_in[0];
  const int*   ei0  = (const int*)d_in[1];
  const int*   ei1  = (const int*)d_in[2];
  const int*   ei2  = (const int*)d_in[3];
  const int*   ei3  = (const int*)d_in[4];
  const float* Wg   = (const float*)d_in[5];
  const float* bg   = (const float*)d_in[6];
  const float* w_ih = (const float*)d_in[7];
  const float* w_hh = (const float*)d_in[8];
  const float* b_ih = (const float*)d_in[9];
  const float* b_hh = (const float*)d_in[10];
  const float* fc_w = (const float*)d_in[11];
  const float* fc_b = (const float*)d_in[12];
  float* out = (float*)d_out;

  char* ws = (char*)d_ws;
  size_t off = 0;
  auto take = [&](size_t bytes) {
    char* p = ws + off;
    off = (off + bytes + 255) & ~(size_t)255;
    return p;
  };
  unsigned short* whh_bf  = (unsigned short*)take(12288ull * 4096 * 2);
  unsigned short* hseq_bf = (unsigned short*)take(192ull * 4096 * 2);
  float*          xp      = (float*)take(192ull * 12288 * 4);
  float* hf[2];
  hf[0] = (float*)take(8ull * 4096 * 4);
  hf[1] = (float*)take(8ull * 4096 * 4);
  unsigned short* hb[2];
  hb[0] = (unsigned short*)take(8ull * 4096 * 2);
  hb[1] = (unsigned short*)take(8ull * 4096 * 2);

  const int n8 = 12288 * 4096 / 8;
  cvt_bf16_kernel<<<(n8 + 255) / 256, 256, 0, stream>>>(w_hh, whh_bf, n8);
  gcn_kernel<<<96, 256, 0, stream>>>(x, ei0, ei1, ei2, ei3, Wg, bg, hseq_bf);
  xp_gemm_kernel<<<dim3(3, 192), 256, 0, stream>>>(hseq_bf, w_ih, b_ih, xp);
  (void)hipMemsetAsync(hf[0], 0, 8 * 4096 * sizeof(float), stream);
  (void)hipMemsetAsync(hb[0], 0, 8 * 4096 * sizeof(unsigned short), stream);
  for (int t = 0; t < 24; ++t) {
    gru_step_kernel<<<256, 96, 0, stream>>>(whh_bf, xp, b_hh,
                                            hf[t & 1], hb[t & 1],
                                            hf[(t + 1) & 1], hb[(t + 1) & 1], t);
  }
  fc_kernel<<<1, 256, 0, stream>>>(hb[0], fc_w, fc_b, out);
  (void)in_sizes; (void)n_in; (void)out_size; (void)ws_size;
}
